// DeltaNet_31877247271522
// MI455X (gfx1250) — matrix operand order repaired and checked by reading, not compiled, not run on hardware
//
#include <hip/hip_runtime.h>
#include <math.h>

#define DI static __device__ __forceinline__

typedef __attribute__((ext_vector_type(16))) __bf16 v16bf;
typedef __attribute__((ext_vector_type(8))) unsigned int v8u;
typedef __attribute__((ext_vector_type(8))) float v8f;

DI unsigned short f2bf(float f) {
  return __builtin_bit_cast(unsigned short, (__bf16)f);  // single v_cvt
}
DI float bf2f(unsigned short s) {
  unsigned int u = ((unsigned int)s) << 16;
  return __builtin_bit_cast(float, u);
}
DI unsigned int pack2bf(float a, float b) {
  return (unsigned int)f2bf(a) | ((unsigned int)f2bf(b) << 16);
}

DI v8f v8zero() {
  v8f z;
#pragma unroll
  for (int i = 0; i < 8; ++i) z[i] = 0.f;
  return z;
}

DI v8f wmma_bf16(v16bf a, v16bf b, v8f c) {
  return __builtin_amdgcn_wmma_f32_16x16x32_bf16(false, a, false, b, (short)0, c,
                                                 false, false);
}

// ---------------- WMMA fragment builders (wave32 layouts per CDNA5 ISA) -------
// A operand (16x32 bf16): lane L -> row M; half h = L>>4; two contiguous K-runs
// of 8 bf16 (16B each): K = k0+h*8+{0..7} and k0+16+h*8+{0..7}.
DI v16bf frag_a_rm(const unsigned short* base, int stride, int M, int k0, int h) {
  const uint4* p = (const uint4*)(base + (size_t)M * stride + k0 + h * 8);
  uint4 lo = p[0], hi = p[2];  // +2 uint4 = +16 ushorts
  v8u t;
  t[0] = lo.x; t[1] = lo.y; t[2] = lo.z; t[3] = lo.w;
  t[4] = hi.x; t[5] = hi.y; t[6] = hi.z; t[7] = hi.w;
  return __builtin_bit_cast(v16bf, t);
}
// B operand (32x16 bf16): lane L -> col N; K = k0+h*8+{0..7} and k0+16+h*8+{0..7}
// (16B aligned).  Array stored [N][Kstride].
DI v16bf frag_b_kmaj(const unsigned short* base, int kstride, int N, int k0, int h) {
  const uint4* p = (const uint4*)(base + (size_t)N * kstride + k0 + h * 8);
  uint4 lo = p[0], hi = p[2];
  v8u t;
  t[0] = lo.x; t[1] = lo.y; t[2] = lo.z; t[3] = lo.w;
  t[4] = hi.x; t[5] = hi.y; t[6] = hi.z; t[7] = hi.w;
  return __builtin_bit_cast(v16bf, t);
}

// ---------------- f32 -> bf16 conversion ----------------
__global__ void k_f32_to_bf16(const float* __restrict__ in,
                              unsigned short* __restrict__ out, int n) {
  int i = blockIdx.x * 256 + threadIdx.x;
  if (i < n) out[i] = f2bf(in[i]);
}

// ---------------- Tiled WMMA GEMM: C[M,N] = A[M,K] @ B[K,N] ----------------
// BM=128, BN=128, BK=32, 8 waves, double-buffered LDS, 1 barrier/iter.
// EPI 0: f32 store.  EPI 1: +bias, exact GELU, bf16 store.
template <int EPI>
__global__ __launch_bounds__(256) void k_gemm_bf16(
    const unsigned short* __restrict__ A, const unsigned short* __restrict__ B,
    const float* __restrict__ bias, void* __restrict__ out, int M, int N, int K) {
  __shared__ unsigned short Al[2][128 * 32];  // [row][k]
  __shared__ unsigned short Bl[2][128 * 32];  // [n][k] K-major
  const int tid = threadIdx.x, lane = tid & 31, wave = tid >> 5;
  const int h = lane >> 4, l15 = lane & 15;
  const int bm = blockIdx.x * 128, bn = blockIdx.y * 128;
  v8f acc[8];
#pragma unroll
  for (int j = 0; j < 8; ++j) acc[j] = v8zero();

  auto load_tiles = [&](int k0, int buf) {
    {  // A tile: 128x32 bf16, 32B per thread
      int row = tid >> 1, seg = tid & 1;
      const uint4* src = (const uint4*)(A + (size_t)(bm + row) * K + k0 + seg * 16);
      uint4* dst = (uint4*)(Al[buf] + row * 32 + seg * 16);
      dst[0] = src[0];
      dst[1] = src[1];
    }
#pragma unroll
    for (int it = 0; it < 2; ++it) {  // B tile: 32x128 -> [n][k]
      int idx = tid + it * 256;       // 512 chunks of 8 bf16
      int k = idx >> 4;
      int n8 = (idx & 15) * 8;
      uint4 bv = *(const uint4*)(B + (size_t)(k0 + k) * N + bn + n8);
      unsigned short us[8];
      *(uint4*)us = bv;
#pragma unroll
      for (int j = 0; j < 8; ++j) Bl[buf][(n8 + j) * 32 + k] = us[j];
    }
  };

  const int nK = K / 32;
  load_tiles(0, 0);
  for (int kk = 0; kk < nK; ++kk) {
    __syncthreads();
    if (kk + 1 < nK) load_tiles((kk + 1) * 32, (kk + 1) & 1);
    if (kk + 2 < nK) {  // prefetch k+2 tiles into L2 (global_prefetch_b8)
      __builtin_prefetch(A + (size_t)(bm + (tid >> 1)) * K + (kk + 2) * 32, 0, 1);
      __builtin_prefetch(B + (size_t)((kk + 2) * 32 + (tid & 31)) * N + bn, 0, 1);
    }
    const int buf = kk & 1;
    v16bf af = frag_a_rm(Al[buf], 32, wave * 16 + l15, 0, h);
#pragma unroll
    for (int j = 0; j < 8; ++j) {
      v16bf bf_ = frag_b_kmaj(Bl[buf], 32, j * 16 + l15, 0, h);
      acc[j] = wmma_bf16(af, bf_, acc[j]);
    }
  }
#pragma unroll
  for (int j = 0; j < 8; ++j) {
#pragma unroll
    for (int r = 0; r < 8; ++r) {
      int row = bm + wave * 16 + r + h * 8;
      int col = bn + j * 16 + l15;
      float x = acc[j][r];
      if (EPI == 1) {
        x += bias[col];
        x = 0.5f * x * (1.f + erff(x * 0.70710678118f));  // exact GELU
        ((unsigned short*)out)[(size_t)row * N + col] = f2bf(x);
      } else {
        ((float*)out)[(size_t)row * N + col] = x;
      }
    }
  }
}

// ---------------- beta = sigmoid(X @ Wb), Wb [1024,4] ----------------
__global__ void k_beta(const float* __restrict__ X, const float* __restrict__ Wb,
                       float* __restrict__ beta) {
  int row = blockIdx.x * 64 + (threadIdx.x >> 2);
  int col = threadIdx.x & 3;
  const float* x = X + (size_t)row * 1024;
  float s = 0.f;
  for (int k = 0; k < 1024; ++k) s += x[k] * Wb[k * 4 + col];
  beta[(size_t)row * 4 + col] = 1.f / (1.f + expf(-s));
}

// ---------------- gate: logits = h1 @ Wf2 + bf2, softmax over groups of 4 ----
__global__ void k_gate(const unsigned short* __restrict__ h1,
                       const float* __restrict__ Wf2, const float* __restrict__ bf2,
                       float* __restrict__ fw) {
  __shared__ float lg[16][16];
  int rr = threadIdx.x >> 4, col = threadIdx.x & 15;
  int row = blockIdx.x * 16 + rr;
  const unsigned short* hr = h1 + (size_t)row * 2048;
  float s = bf2[col];
  for (int k = 0; k < 2048; ++k) s += bf2f(hr[k]) * Wf2[k * 16 + col];
  lg[rr][col] = s;
  __syncthreads();
  int g = col & ~3;
  float m = lg[rr][g];
#pragma unroll
  for (int j = 1; j < 4; ++j) m = fmaxf(m, lg[rr][g + j]);
  float den = 0.f;
#pragma unroll
  for (int j = 0; j < 4; ++j) den += expf(lg[rr][g + j] - m);
  fw[(size_t)row * 16 + col] = expf(s - m) / den;
}

// ---------------- depthwise causal conv K=4 + SiLU ----------------
__global__ void k_conv4_silu(const float* __restrict__ x, const float* __restrict__ w,
                             float* __restrict__ y) {
  int idx = blockIdx.x * 256 + threadIdx.x;
  int c = idx & 1023;
  int bl = idx >> 10;
  int l = bl & 4095;
  float s = 0.f;
#pragma unroll
  for (int t = 0; t < 4; ++t) {
    int lt = l - 3 + t;
    if (lt >= 0) s += x[(size_t)(bl - 3 + t) * 1024 + c] * w[c * 4 + t];
  }
  y[idx] = s / (1.f + expf(-s));  // silu
}

// ---------------- FIR short(7) + long(64) depthwise causal ----------------
__global__ void k_fir(const float* __restrict__ v, const float* __restrict__ w7,
                      const float* __restrict__ w64, float* __restrict__ ls,
                      float* __restrict__ ll) {
  int idx = blockIdx.x * 256 + threadIdx.x;
  int c = idx & 1023;
  int bl = idx >> 10;
  int l = bl & 4095;
  float a = 0.f;
#pragma unroll
  for (int t = 0; t < 7; ++t) {
    int lt = l - 6 + t;
    if (lt >= 0) a += v[(size_t)(bl - 6 + t) * 1024 + c] * w7[c * 7 + t];
  }
  float bsum = 0.f;
  for (int t = 0; t < 64; ++t) {
    int lt = l - 63 + t;
    if (lt >= 0) bsum += v[(size_t)(bl - 63 + t) * 1024 + c] * w64[c * 64 + t];
  }
  ls[idx] = a;
  ll[idx] = bsum;
}

// ---------------- fused combine + RMSNorm -> bf16 ----------------
__global__ __launch_bounds__(256) void k_combine(
    const float* __restrict__ v, const float* __restrict__ ls,
    const float* __restrict__ ll, const float* __restrict__ dl,
    const float* __restrict__ fw, const float* __restrict__ rmsw,
    unsigned short* __restrict__ onorm) {
  int lane = threadIdx.x & 31, wave = threadIdx.x >> 5;
  int item = blockIdx.x * 8 + wave;  // (bl, h) per wave
  int h = item & 3, bl = item >> 2;
  const float* f = fw + (size_t)bl * 16 + h * 4;
  float f0 = f[0], f1 = f[1], f2 = f[2], f3 = f[3];
  size_t base = (size_t)bl * 1024 + h * 256;
  float o[8], ss = 0.f;
#pragma unroll
  for (int i = 0; i < 8; ++i) {
    size_t ix = base + i * 32 + lane;
    float val = f0 * ls[ix] + f1 * ll[ix] + f2 * dl[ix] + f3 * v[ix];
    o[i] = val;
    ss += val * val;
  }
#pragma unroll
  for (int m = 16; m > 0; m >>= 1) ss += __shfl_xor(ss, m, 32);
  float rs = rsqrtf(ss * (1.f / 256.f) + 1e-5f);
#pragma unroll
  for (int i = 0; i < 8; ++i) {
    int d = i * 32 + lane;
    onorm[base + d] = f2bf(o[i] * rs * rmsw[d]);
  }
}

// ---------------- chunkwise delta rule ----------------
// One block per (b, h, dv-slice): 8 heads x 4 slices of 64 dv = 32 blocks.
// S slice [64 e][256 d] kept in global f32 + bf16 mirror (L2-resident).
// T = -(k_beta @ k^T) = -diag(beta) * (k @ k^T): compute symmetric k@k^T with
// contiguous fragments and apply beta row-scale at f32 write-out.
__global__ __launch_bounds__(256) void k_delta(
    const float* __restrict__ qg, const float* __restrict__ kg,
    const float* __restrict__ vg, const float* __restrict__ betag,
    float* __restrict__ outp, float* __restrict__ Sall,
    unsigned short* __restrict__ S16all, unsigned short* __restrict__ qnall,
    unsigned short* __restrict__ kTall, float* __restrict__ uall) {
  __shared__ unsigned short BK[32 * 256];    // normalized k, row-major [r][d]
  __shared__ unsigned short BKBT[256 * 32];  // k_beta^T [d][r]; later u_t^T [e][r]
  __shared__ unsigned short BVT[256 * 32];   // v_beta^T slice [e<64][r]; later w [r][d]
  __shared__ float Tf[32 * 32];
  __shared__ unsigned short T16[32 * 32];
  __shared__ unsigned short attn16[32 * 32];
  __shared__ float betaRow[32];

  const int tid = threadIdx.x, lane = tid & 31, wave = tid >> 5;
  const int h16 = lane >> 4, l15 = lane & 15;
  const int bhs = blockIdx.x;
  const int sl = bhs & 3, bh = bhs >> 2;
  const int b = bh >> 2, hh = bh & 3;
  const int e0g = sl * 64;                           // dv slice base
  float* S = Sall + (size_t)bhs * 16384;             // S^T slice [e<64][d<256] f32
  unsigned short* S16 = S16all + (size_t)bhs * 16384;
  unsigned short* qn = qnall + (size_t)bhs * 8192;   // normalized q [r][d]
  unsigned short* kT = kTall + (size_t)bhs * 8192;   // k^T [d][r]
  float* uT = uall + (size_t)bhs * 2048;             // u^T slice [e<64][r]

  for (int i = tid; i < 16384; i += 256) {
    S[i] = 0.f;
    S16[i] = 0;
  }
  __threadfence();
  __syncthreads();

  for (int n = 0; n < 128; ++n) {
    // ---- step 1: load rows, l2norm q/k, beta scale, stage tiles ----
    for (int rr = 0; rr < 4; ++rr) {
      int r = wave * 4 + rr;
      int l = n * 32 + r;
      size_t rowbase = ((size_t)(b * 4096 + l)) * 1024 + hh * 256;
      float4 q0 = *(const float4*)(qg + rowbase + lane * 8);
      float4 q1 = *(const float4*)(qg + rowbase + lane * 8 + 4);
      float4 k0 = *(const float4*)(kg + rowbase + lane * 8);
      float4 k1 = *(const float4*)(kg + rowbase + lane * 8 + 4);
      float sq = q0.x * q0.x + q0.y * q0.y + q0.z * q0.z + q0.w * q0.w +
                 q1.x * q1.x + q1.y * q1.y + q1.z * q1.z + q1.w * q1.w;
      float sk = k0.x * k0.x + k0.y * k0.y + k0.z * k0.z + k0.w * k0.w +
                 k1.x * k1.x + k1.y * k1.y + k1.z * k1.z + k1.w * k1.w;
#pragma unroll
      for (int m = 16; m > 0; m >>= 1) {
        sq += __shfl_xor(sq, m, 32);
        sk += __shfl_xor(sk, m, 32);
      }
      float rq = rsqrtf(sq), rk = rsqrtf(sk);
      float bet = betag[(size_t)(b * 4096 + l) * 4 + hh];
      if (lane == 0) betaRow[r] = bet;
      // packed bf16 row stores for qn / kn (16B each)
      uint4 qp, kp;
      qp.x = pack2bf(q0.x * rq, q0.y * rq);
      qp.y = pack2bf(q0.z * rq, q0.w * rq);
      qp.z = pack2bf(q1.x * rq, q1.y * rq);
      qp.w = pack2bf(q1.z * rq, q1.w * rq);
      kp.x = pack2bf(k0.x * rk, k0.y * rk);
      kp.y = pack2bf(k0.z * rk, k0.w * rk);
      kp.z = pack2bf(k1.x * rk, k1.y * rk);
      kp.w = pack2bf(k1.z * rk, k1.w * rk);
      *(uint4*)(qn + r * 256 + lane * 8) = qp;
      *(uint4*)(BK + r * 256 + lane * 8) = kp;
      // transposed (strided) stores for k^T and k_beta^T
      float kf[8] = {k0.x, k0.y, k0.z, k0.w, k1.x, k1.y, k1.z, k1.w};
#pragma unroll
      for (int j = 0; j < 8; ++j) {
        int d = lane * 8 + j;
        kT[d * 32 + r] = f2bf(kf[j] * rk);
        BKBT[d * 32 + r] = f2bf(kf[j] * rk * bet);
      }
      // v slice: 64 dv values per row
      float2 vv = *(const float2*)(vg + rowbase + e0g + lane * 2);
      BVT[(lane * 2 + 0) * 32 + r] = f2bf(vv.x * bet);
      BVT[(lane * 2 + 1) * 32 + r] = f2bf(vv.y * bet);
    }
    __threadfence();
    __syncthreads();

    // ---- step 2: T = -diag(beta) (k @ k^T), strictly lower (4 waves) ----
    if (wave < 4) {
      int m0 = (wave >> 1) * 16, n0 = (wave & 1) * 16;
      v8f acc = v8zero();
      for (int kk = 0; kk < 8; ++kk) {
        v16bf a = frag_a_rm(BK, 256, m0 + l15, kk * 32, h16);
        v16bf bb = frag_b_kmaj(BK, 256, n0 + l15, kk * 32, h16);
        acc = wmma_bf16(a, bb, acc);
      }
#pragma unroll
      for (int r8 = 0; r8 < 8; ++r8) {
        int i = m0 + r8 + h16 * 8, j = n0 + l15;
        Tf[i * 32 + j] = (j < i) ? -betaRow[i] * acc[r8] : 0.f;
      }
    }
    __syncthreads();

    // ---- step 3: forward substitution (sequential, wave 0), T += I ----
    if (wave == 0) {
      int j = lane;
      for (int i = 1; i < 32; ++i) {
        float s = 0.f;
        for (int m = 0; m < i; ++m) s += Tf[i * 32 + m] * Tf[m * 32 + j];
        Tf[i * 32 + j] += s;
      }
      Tf[j * 32 + j] += 1.f;
    }
    __syncthreads();
    for (int i = tid; i < 512; i += 256)
      *(unsigned int*)(T16 + 2 * i) = pack2bf(Tf[2 * i], Tf[2 * i + 1]);
    __syncthreads();

    // ---- step 3a: u = T @ v_beta (slice) -> uT [e][r] (1 tile/wave) ----
    {
      int m0 = (wave >> 2) * 16, e0 = (wave & 3) * 16;
      v16bf a = frag_a_rm(T16, 32, m0 + l15, 0, h16);
      v16bf bb = frag_b_kmaj(BVT, 32, e0 + l15, 0, h16);
      v8f acc = wmma_bf16(a, bb, v8zero());
#pragma unroll
      for (int r8 = 0; r8 < 8; ++r8)
        uT[(e0 + l15) * 32 + m0 + r8 + h16 * 8] = acc[r8];
    }
    __threadfence();
    __syncthreads();

    // ---- step 3b: w = T @ k_beta (full dk) -> row-major into BVT ----
    for (int it = 0; it < 4; ++it) {
      int t = wave * 4 + it;
      int m0 = (t >> 4) * 16, d0 = (t & 15) * 16;
      v16bf a = frag_a_rm(T16, 32, m0 + l15, 0, h16);
      v16bf bb = frag_b_kmaj(BKBT, 32, d0 + l15, 0, h16);
      v8f acc = wmma_bf16(a, bb, v8zero());
#pragma unroll
      for (int r8 = 0; r8 < 8; ++r8)
        BVT[(m0 + r8 + h16 * 8) * 256 + d0 + l15] = f2bf(acc[r8]);
    }
    __syncthreads();

    // ---- step 4+5: u_t = u - w @ S -> u_t^T bf16 into BKBT (1 tile/wave) ----
    {
      int m0 = (wave >> 2) * 16, e0 = (wave & 3) * 16;
      v8f acc = v8zero();
      for (int kk = 0; kk < 8; ++kk) {
        v16bf a = frag_a_rm(BVT, 256, m0 + l15, kk * 32, h16);  // w
        v16bf bb = frag_b_kmaj(S16, 256, e0 + l15, kk * 32, h16);
        acc = wmma_bf16(a, bb, acc);
      }
#pragma unroll
      for (int r8 = 0; r8 < 8; ++r8) {
        int r = m0 + r8 + h16 * 8, e = e0 + l15;
        BKBT[e * 32 + r] = f2bf(uT[e * 32 + r] - acc[r8]);
      }
    }
    __syncthreads();

    // ---- step 6: attn = (q @ k^T) * causal (incl. diagonal) ----
    if (wave < 4) {
      int m0 = (wave >> 1) * 16, n0 = (wave & 1) * 16;
      v8f acc = v8zero();
      for (int kk = 0; kk < 8; ++kk) {
        v16bf a = frag_a_rm(qn, 256, m0 + l15, kk * 32, h16);
        v16bf bb = frag_b_kmaj(BK, 256, n0 + l15, kk * 32, h16);
        acc = wmma_bf16(a, bb, acc);
      }
#pragma unroll
      for (int r8 = 0; r8 < 8; ++r8) {
        int i = m0 + r8 + h16 * 8, j = n0 + l15;
        attn16[i * 32 + j] = f2bf((j <= i) ? acc[r8] : 0.f);
      }
    }
    __syncthreads();

    // ---- step 7: o = q @ S + attn @ u_t (1 tile/wave) ----
    {
      int m0 = (wave >> 2) * 16, e0 = (wave & 3) * 16;
      v8f acc = v8zero();
      for (int kk = 0; kk < 8; ++kk) {
        v16bf a = frag_a_rm(qn, 256, m0 + l15, kk * 32, h16);
        v16bf bb = frag_b_kmaj(S16, 256, e0 + l15, kk * 32, h16);
        acc = wmma_bf16(a, bb, acc);
      }
      v16bf a = frag_a_rm(attn16, 32, m0 + l15, 0, h16);
      v16bf bb = frag_b_kmaj(BKBT, 32, e0 + l15, 0, h16);
      acc = wmma_bf16(a, bb, acc);
#pragma unroll
      for (int r8 = 0; r8 < 8; ++r8) {
        int r = m0 + r8 + h16 * 8;
        int l = n * 32 + r;
        outp[((size_t)(b * 4096 + l) * 4 + hh) * 256 + e0g + e0 + l15] = acc[r8];
      }
    }
    __syncthreads();

    // ---- step 8: S += k^T @ u_t (64 tiles, 8/wave, RMW via C operand) ----
    for (int it = 0; it < 8; ++it) {
      int t = wave * 8 + it;
      int d0 = (t >> 2) * 16, e0 = (t & 3) * 16;
      v16bf a = frag_a_rm(kT, 32, d0 + l15, 0, h16);
      v16bf bb = frag_b_kmaj(BKBT, 32, e0 + l15, 0, h16);
      v8f c;
#pragma unroll
      for (int r8 = 0; r8 < 8; ++r8)
        c[r8] = S[(size_t)(e0 + l15) * 256 + d0 + r8 + h16 * 8];
      c = wmma_bf16(a, bb, c);
#pragma unroll
      for (int r8 = 0; r8 < 8; ++r8) {
        size_t ix = (size_t)(e0 + l15) * 256 + d0 + r8 + h16 * 8;
        S[ix] = c[r8];
        S16[ix] = f2bf(c[r8]);
      }
    }
    __threadfence();
    __syncthreads();
  }
}

// ======================= host launcher =======================
extern "C" void kernel_launch(void* const* d_in, const int* in_sizes, int n_in,
                              void* d_out, int out_size, void* d_ws, size_t ws_size,
                              hipStream_t stream) {
  (void)in_sizes; (void)n_in; (void)out_size; (void)ws_size;
  const float* hidden = (const float*)d_in[0];
  const float* Wq = (const float*)d_in[1];
  const float* Wk = (const float*)d_in[2];
  const float* Wv = (const float*)d_in[3];
  const float* Wb = (const float*)d_in[4];
  const float* cq = (const float*)d_in[5];
  const float* ck = (const float*)d_in[6];
  const float* cv = (const float*)d_in[7];
  const float* fs = (const float*)d_in[8];
  const float* fl = (const float*)d_in[9];
  const float* Wf1 = (const float*)d_in[10];
  const float* bf1 = (const float*)d_in[11];
  const float* Wf2 = (const float*)d_in[12];
  const float* bf2 = (const float*)d_in[13];
  const float* rmsw = (const float*)d_in[14];
  const float* Wo = (const float*)d_in[15];

  const int BL = 8192, H = 1024;
  char* base = (char*)d_ws;
  size_t off = 0;
  auto alloc = [&](size_t bytes) -> void* {
    void* p = base + off;
    off += (bytes + 255) & ~(size_t)255;
    return p;
  };
  unsigned short* hb = (unsigned short*)alloc((size_t)BL * H * 2);
  unsigned short* wqb = (unsigned short*)alloc((size_t)H * H * 2);
  unsigned short* wkb = (unsigned short*)alloc((size_t)H * H * 2);
  unsigned short* wvb = (unsigned short*)alloc((size_t)H * H * 2);
  unsigned short* wf1b = (unsigned short*)alloc((size_t)H * 2048 * 2);
  unsigned short* wob = (unsigned short*)alloc((size_t)H * H * 2);
  float* qpre = (float*)alloc((size_t)BL * H * 4);
  float* kpre = (float*)alloc((size_t)BL * H * 4);
  float* vpre = (float*)alloc((size_t)BL * H * 4);
  float* qcv = (float*)alloc((size_t)BL * H * 4);
  float* kcv = (float*)alloc((size_t)BL * H * 4);
  float* vcv = (float*)alloc((size_t)BL * H * 4);
  float* beta = (float*)alloc((size_t)BL * 4 * 4);
  unsigned short* h1 = (unsigned short*)alloc((size_t)BL * 2048 * 2);
  float* fw = (float*)alloc((size_t)BL * 16 * 4);
  float* Sws = (float*)alloc((size_t)32 * 16384 * 4);
  unsigned short* S16ws = (unsigned short*)alloc((size_t)32 * 16384 * 2);
  unsigned short* qnws = (unsigned short*)alloc((size_t)32 * 8192 * 2);
  unsigned short* kTws = (unsigned short*)alloc((size_t)32 * 8192 * 2);
  float* uws = (float*)alloc((size_t)32 * 2048 * 4);
  // buffer reuse (stream-ordered, disjoint live ranges):
  float* lsb = qpre;           // FIR short (qpre consumed by conv_q)
  float* llb = kpre;           // FIR long
  float* dlt = vpre;           // delta_out (vpre consumed by conv_v)
  unsigned short* onorm = h1;  // RMSNorm'ed output (h1 consumed by gate)

  k_f32_to_bf16<<<(BL * H + 255) / 256, 256, 0, stream>>>(hidden, hb, BL * H);
  k_f32_to_bf16<<<(H * H + 255) / 256, 256, 0, stream>>>(Wq, wqb, H * H);
  k_f32_to_bf16<<<(H * H + 255) / 256, 256, 0, stream>>>(Wk, wkb, H * H);
  k_f32_to_bf16<<<(H * H + 255) / 256, 256, 0, stream>>>(Wv, wvb, H * H);
  k_f32_to_bf16<<<(H * 2048 + 255) / 256, 256, 0, stream>>>(Wf1, wf1b, H * 2048);
  k_f32_to_bf16<<<(H * H + 255) / 256, 256, 0, stream>>>(Wo, wob, H * H);

  dim3 g1(BL / 128, H / 128);
  k_gemm_bf16<0><<<g1, 256, 0, stream>>>(hb, wqb, nullptr, qpre, BL, H, H);
  k_gemm_bf16<0><<<g1, 256, 0, stream>>>(hb, wkb, nullptr, kpre, BL, H, H);
  k_gemm_bf16<0><<<g1, 256, 0, stream>>>(hb, wvb, nullptr, vpre, BL, H, H);
  k_beta<<<BL / 64, 256, 0, stream>>>(hidden, Wb, beta);
  dim3 g2(BL / 128, 2048 / 128);
  k_gemm_bf16<1><<<g2, 256, 0, stream>>>(hb, wf1b, bf1, h1, BL, 2048, H);
  k_gate<<<BL / 16, 256, 0, stream>>>(h1, Wf2, bf2, fw);

  k_conv4_silu<<<BL * H / 256, 256, 0, stream>>>(qpre, cq, qcv);
  k_conv4_silu<<<BL * H / 256, 256, 0, stream>>>(kpre, ck, kcv);
  k_conv4_silu<<<BL * H / 256, 256, 0, stream>>>(vpre, cv, vcv);
  k_fir<<<BL * H / 256, 256, 0, stream>>>(vcv, fs, fl, lsb, llb);

  k_delta<<<32, 256, 0, stream>>>(qcv, kcv, vcv, beta, dlt, Sws, S16ws, qnws,
                                  kTws, uws);

  k_combine<<<BL * 4 / 8, 256, 0, stream>>>(vcv, lsb, llb, dlt, fw, rmsw, onorm);
  k_gemm_bf16<0><<<g1, 256, 0, stream>>>(onorm, wob, nullptr, d_out, BL, H, H);
}
